// MultiHeadAttention_89343909691687
// MI455X (gfx1250) — matrix operand order repaired and checked by reading, not compiled, not run on hardware
//
#include <hip/hip_runtime.h>
#include <cstdint>
#include <cstddef>

// ---------------- problem constants ----------------
namespace {
constexpr int Bc  = 4;
constexpr int Tc  = 1024;
constexpr int Dc  = 1024;
constexpr int Hc  = 16;
constexpr int DKc = 64;
constexpr int BHc = Bc * Hc;   // 64
constexpr int Mr  = Bc * Tc;   // 4096 rows for the projections
}

typedef __attribute__((ext_vector_type(16))) __bf16 v16bf;
typedef __attribute__((ext_vector_type(8)))  float  v8f;

// ---------------- WMMA fragment helpers (CDNA5 16x16x32 bf16) ----------------
// A fragment (16x32, row-major source, row stride ld):
//   lane: row = lane%16, half = lane/16
//   a[0..7]  = A[row][half*8 + j],  a[8..15] = A[row][16 + half*8 + j]
__device__ __forceinline__ v16bf load_a_bf16(const __bf16* src, int ld, int lane) {
  int row = lane & 15, half = lane >> 4;
  v16bf a;
#pragma unroll
  for (int j = 0; j < 8; ++j) {
    a[j]     = src[row * ld + half * 8 + j];
    a[j + 8] = src[row * ld + 16 + half * 8 + j];
  }
  return a;
}

// B fragment (32x16), source stored row-major N x K (element (k,n) = src[n*ld + k]):
//   lane: col = lane%16, half = lane/16;  b[0..7] = B[half*8 + j][col], b[8..15] = B[16 + half*8 + j][col]
// -> 2x 16 contiguous bytes per lane (2x ds_load_b128)
__device__ __forceinline__ v16bf load_bT_bf16(const __bf16* src, int ld, int lane) {
  int col = lane & 15, half = lane >> 4;
  v16bf b;
#pragma unroll
  for (int j = 0; j < 8; ++j) {
    b[j]     = src[col * ld + half * 8 + j];
    b[j + 8] = src[col * ld + 16 + half * 8 + j];
  }
  return b;
}

__device__ __forceinline__ v8f wmma_bf16(v16bf a, v16bf b, v8f c) {
  return __builtin_amdgcn_wmma_f32_16x16x32_bf16(false, a, false, b, (short)0, c,
                                                 false, false);
}

// ---------------- fp32 -> bf16 conversion ----------------
__global__ void cvt_f32_bf16_kernel(const float* __restrict__ in,
                                    __bf16* __restrict__ out, int n) {
  int i = blockIdx.x * blockDim.x + threadIdx.x;
  int stride = gridDim.x * blockDim.x;
  for (; i < n; i += stride) out[i] = (__bf16)in[i];
}

// ---------------- LayerNorm (one block per row of 1024) ----------------
__global__ __launch_bounds__(256)
void layernorm_kernel(const float* __restrict__ x, const float* __restrict__ g,
                      const float* __restrict__ be, __bf16* __restrict__ xn) {
  __shared__ float rs[8], rss[8];
  int tid = threadIdx.x, wave = tid >> 5, lane = tid & 31;
  size_t base = (size_t)blockIdx.x * Dc;
  float v[4], s = 0.f, ss = 0.f;
#pragma unroll
  for (int i = 0; i < 4; ++i) {
    v[i] = x[base + i * 256 + tid];
    s += v[i];
    ss += v[i] * v[i];
  }
#pragma unroll
  for (int off = 16; off; off >>= 1) {
    s += __shfl_xor(s, off, 32);
    ss += __shfl_xor(ss, off, 32);
  }
  if (lane == 0) { rs[wave] = s; rss[wave] = ss; }
  __syncthreads();
  s = 0.f; ss = 0.f;
#pragma unroll
  for (int i = 0; i < 8; ++i) { s += rs[i]; ss += rss[i]; }
  float mu = s * (1.0f / Dc);
  float var = ss * (1.0f / Dc) - mu * mu;
  float rsig = rsqrtf(var + 1e-5f);
#pragma unroll
  for (int i = 0; i < 4; ++i) {
    int c = i * 256 + tid;
    xn[base + c] = (__bf16)((v[i] - mu) * rsig * g[c] + be[c]);
  }
}

// ---------------- Generic GEMM: C = A(MxK) @ Bw(NxK)^T + bias ----------------
// A,Bw bf16 row-major. Block tile 128(M) x 128(N), K-step 64.
// 8 waves: wave grid 4m x 2n; per wave 2 m-tiles x 4 n-tiles (16 WMMA / stage).
template <bool F32OUT>
__global__ __launch_bounds__(256)
void gemm_bf16_kernel(const __bf16* __restrict__ A, const __bf16* __restrict__ Bw,
                      const float* __restrict__ bias, void* __restrict__ outp,
                      int M, int N, int K) {
  constexpr int BM = 128, BN = 128, BK = 64, LDA = 72, LDB = 72;
  __shared__ __bf16 As[BM * LDA];
  __shared__ __bf16 Bs[BN * LDB];
  int tid = threadIdx.x, wave = tid >> 5, lane = tid & 31;
  int m0 = blockIdx.x * BM, n0 = blockIdx.y * BN;
  int wm = wave & 3, wn = wave >> 2;

  v8f acc[2][4] = {};

  for (int kk = 0; kk < K; kk += BK) {
    {  // stage A and B tiles: each thread copies 64B per tile (4x b128)
      int row = tid >> 1, hh = tid & 1;
      const __bf16* srcA = A + (size_t)(m0 + row) * K + kk + hh * 32;
      const __bf16* srcB = Bw + (size_t)(n0 + row) * K + kk + hh * 32;
      __bf16* dstA = &As[row * LDA + hh * 32];
      __bf16* dstB = &Bs[row * LDB + hh * 32];
#pragma unroll
      for (int j = 0; j < 4; ++j) {
        ((uint4*)dstA)[j] = ((const uint4*)srcA)[j];
        ((uint4*)dstB)[j] = ((const uint4*)srcB)[j];
      }
    }
    __syncthreads();

#pragma unroll
    for (int kc = 0; kc < 2; ++kc) {
      v16bf af[2], bfr[4];
#pragma unroll
      for (int mt = 0; mt < 2; ++mt)
        af[mt] = load_a_bf16(&As[(wm * 32 + mt * 16) * LDA + kc * 32], LDA, lane);
#pragma unroll
      for (int nt = 0; nt < 4; ++nt)
        bfr[nt] = load_bT_bf16(&Bs[(wn * 64 + nt * 16) * LDB + kc * 32], LDB, lane);
#pragma unroll
      for (int mt = 0; mt < 2; ++mt)
#pragma unroll
        for (int nt = 0; nt < 4; ++nt)
          acc[mt][nt] = wmma_bf16(af[mt], bfr[nt], acc[mt][nt]);
    }
    __syncthreads();
  }

  int half = lane >> 4, coln = lane & 15;
#pragma unroll
  for (int mt = 0; mt < 2; ++mt)
#pragma unroll
    for (int nt = 0; nt < 4; ++nt) {
      int mbase = m0 + wm * 32 + mt * 16 + 8 * half;
      int n = n0 + wn * 64 + nt * 16 + coln;
      float bn = bias[n];
#pragma unroll
      for (int r = 0; r < 8; ++r) {
        float v = acc[mt][nt][r] + bn;
        size_t idx = (size_t)(mbase + r) * N + n;
        if (F32OUT) ((float*)outp)[idx] = v;
        else        ((__bf16*)outp)[idx] = (__bf16)v;
      }
    }
}

// ---------------- Relative-position bias ----------------
// bias[bh][qi][kj] = sum_d pos_k[qi][kj][d] * q[b,h,qi,d]   (batched over bh)
// Per block: fixed qi, 128 kj rows; Qg staged in LDS as [bh][d] (row-major NxK
// so B-fragments are contiguous ds_load_b128); pos_k fp32 streamed exactly once
// device-wide and converted to bf16 on the fly while building A-fragments.
__global__ __launch_bounds__(256)
void bias_kernel(const __bf16* __restrict__ qb, const float* __restrict__ posk,
                 __bf16* __restrict__ biasb) {
  constexpr int LDQ = 72;
  __shared__ __bf16 Qs[64 * LDQ];  // [bh][d]
  int tid = threadIdx.x, wave = tid >> 5, lane = tid & 31;
  int qi = blockIdx.y, kb0 = blockIdx.x * 128;

  {  // stage Qg: thread copies 16 contiguous d for one bh (2x b128)
    int bhp = tid >> 2, c = tid & 3;
    const __bf16* src =
        qb + ((size_t)((bhp >> 4) * Tc + qi)) * Dc + (bhp & 15) * DKc + c * 16;
    __bf16* dst = &Qs[bhp * LDQ + c * 16];
    ((uint4*)dst)[0] = ((const uint4*)src)[0];
    ((uint4*)dst)[1] = ((const uint4*)src)[1];
  }
  __syncthreads();

  int kj0 = kb0 + wave * 16;
  int rowl = lane & 15, half = lane >> 4, coln = lane & 15;
  v8f acc[4] = {};
#pragma unroll
  for (int kc = 0; kc < 2; ++kc) {
    const float* srcA = posk + ((size_t)qi * Tc + kj0 + rowl) * DKc + kc * 32;
    v16bf a;
#pragma unroll
    for (int j = 0; j < 8; ++j) {
      a[j]     = (__bf16)srcA[half * 8 + j];
      a[j + 8] = (__bf16)srcA[16 + half * 8 + j];
    }
#pragma unroll
    for (int nt = 0; nt < 4; ++nt) {
      v16bf bq = load_bT_bf16(&Qs[(nt * 16) * LDQ + kc * 32], LDQ, lane);
      acc[nt] = wmma_bf16(a, bq, acc[nt]);
    }
  }
#pragma unroll
  for (int nt = 0; nt < 4; ++nt)
#pragma unroll
    for (int r = 0; r < 8; ++r) {
      int bh = nt * 16 + coln;
      int kj = kj0 + r + 8 * half;
      biasb[(size_t)bh * Tc * Tc + (size_t)qi * Tc + kj] = (__bf16)acc[nt][r];
    }
}

// ---------------- Flash attention (per (b,h), 16 q-rows per wave) ----------------
__global__ __launch_bounds__(256)
void attn_kernel(const __bf16* __restrict__ qb, const __bf16* __restrict__ kb,
                 const __bf16* __restrict__ vb, const __bf16* __restrict__ biasb,
                 __bf16* __restrict__ outb) {
  constexpr int CH = 32, LDK = 72, LDVT = 40, LDP = 40;
  __shared__ __bf16 Ks[CH * LDK];        // [key][d]
  __shared__ __bf16 Vt[DKc * LDVT];      // [d][key]  (transposed for bT frags)
  __shared__ __bf16 Ps[8][16 * LDP];     // per-wave P slab

  int tid = threadIdx.x, wave = tid >> 5, lane = tid & 31;
  int bh = blockIdx.y, b = bh >> 4, h = bh & 15;
  int q0 = blockIdx.x * 128 + wave * 16;
  int rowl = lane & 15, half = lane >> 4, coln = lane & 15;

  // Q fragments for this wave's 16 rows (K = 64 -> two 16x32 A-frags)
  const __bf16* qbase = qb + ((size_t)(b * Tc + q0)) * Dc + h * DKc;
  v16bf qa[2];
#pragma unroll
  for (int kc = 0; kc < 2; ++kc) {
#pragma unroll
    for (int j = 0; j < 8; ++j) {
      qa[kc][j]     = qbase[rowl * Dc + kc * 32 + half * 8 + j];
      qa[kc][j + 8] = qbase[rowl * Dc + kc * 32 + 16 + half * 8 + j];
    }
  }

  float m[8], l[8];
  v8f accO[4] = {};
#pragma unroll
  for (int r = 0; r < 8; ++r) { m[r] = -3.0e38f; l[r] = 0.f; }

  const float SC = 0.125f * 1.44269504088896f;  // 1/sqrt(64) * log2(e)
  const __bf16* bias_bh = biasb + (size_t)bh * Tc * Tc;

  for (int kt = 0; kt < Tc; kt += CH) {
    __syncthreads();
    {  // stage K chunk [key][d] and V chunk transposed [d][key]
      int key = tid >> 3, c = tid & 7;
      size_t rowoff = ((size_t)(b * Tc + kt + key)) * Dc + h * DKc + c * 8;
      *(uint4*)&Ks[key * LDK + c * 8] = *(const uint4*)(kb + rowoff);
      union { uint4 u; __bf16 e[8]; } tv;
      tv.u = *(const uint4*)(vb + rowoff);
#pragma unroll
      for (int j = 0; j < 8; ++j) Vt[(c * 8 + j) * LDVT + key] = tv.e[j];
    }
    __syncthreads();
    if (kt + CH < Tc)
      __builtin_prefetch(kb + ((size_t)(b * Tc + kt + CH)) * Dc + h * DKc, 0, 0);

    // S = Q * K^T  (16 x 32, two 16-key N-tiles)
    v8f s[2] = {};
#pragma unroll
    for (int nt = 0; nt < 2; ++nt)
#pragma unroll
      for (int kc = 0; kc < 2; ++kc) {
        v16bf bK = load_bT_bf16(&Ks[(nt * 16) * LDK + kc * 32], LDK, lane);
        s[nt] = wmma_bf16(qa[kc], bK, s[nt]);
      }

    // online softmax (rows live in 16-lane groups of one VGPR)
#pragma unroll
    for (int r = 0; r < 8; ++r) {
      int qrow = q0 + r + 8 * half;
      float t0 = (s[0][r] + (float)bias_bh[(size_t)qrow * Tc + kt + coln]) * SC;
      float t1 = (s[1][r] + (float)bias_bh[(size_t)qrow * Tc + kt + 16 + coln]) * SC;
      float mx = fmaxf(t0, t1);
#pragma unroll
      for (int off = 1; off < 16; off <<= 1) mx = fmaxf(mx, __shfl_xor(mx, off, 32));
      float mn = fmaxf(m[r], mx);
      float alpha = exp2f(m[r] - mn);
      float p0 = exp2f(t0 - mn), p1 = exp2f(t1 - mn);
      float ps = p0 + p1;
#pragma unroll
      for (int off = 1; off < 16; off <<= 1) ps += __shfl_xor(ps, off, 32);
      l[r] = l[r] * alpha + ps;
      m[r] = mn;
      accO[0][r] *= alpha; accO[1][r] *= alpha;
      accO[2][r] *= alpha; accO[3][r] *= alpha;
      Ps[wave][(r + 8 * half) * LDP + coln]      = (__bf16)p0;
      Ps[wave][(r + 8 * half) * LDP + 16 + coln] = (__bf16)p1;
    }

    // O += P * V   (P re-fragmented via per-wave LDS slab; V from transposed slab)
    v16bf pa = load_a_bf16(&Ps[wave][0], LDP, lane);
#pragma unroll
    for (int dt = 0; dt < 4; ++dt) {
      v16bf bV = load_bT_bf16(&Vt[(dt * 16) * LDVT], LDVT, lane);
      accO[dt] = wmma_bf16(pa, bV, accO[dt]);
    }
  }

  // normalize and write (b,t,h,dk) bf16
#pragma unroll
  for (int dt = 0; dt < 4; ++dt)
#pragma unroll
    for (int r = 0; r < 8; ++r) {
      int qrow = q0 + r + 8 * half;
      float o = accO[dt][r] / l[r];
      outb[((size_t)(b * Tc + qrow)) * Dc + h * DKc + dt * 16 + coln] = (__bf16)o;
    }
}

// ---------------- launcher ----------------
extern "C" void kernel_launch(void* const* d_in, const int* in_sizes, int n_in,
                              void* d_out, int out_size, void* d_ws, size_t ws_size,
                              hipStream_t stream) {
  (void)in_sizes; (void)n_in; (void)out_size; (void)ws_size;
  const float* x     = (const float*)d_in[0];
  const float* pos_k = (const float*)d_in[1];
  const float* ln_g  = (const float*)d_in[2];
  const float* ln_b  = (const float*)d_in[3];
  const float* Wq    = (const float*)d_in[4];
  const float* bq    = (const float*)d_in[5];
  const float* Wk    = (const float*)d_in[6];
  const float* bk    = (const float*)d_in[7];
  const float* Wv    = (const float*)d_in[8];
  const float* bv    = (const float*)d_in[9];
  const float* Wo    = (const float*)d_in[10];
  const float* bo    = (const float*)d_in[11];
  float* out = (float*)d_out;

  char* ws = (char*)d_ws;
  size_t off = 0;
  auto alloc = [&](size_t bytes) -> void* {
    void* p = ws + off;
    off += (bytes + 255) & ~(size_t)255;
    return p;
  };
  __bf16* xn    = (__bf16*)alloc((size_t)Mr * Dc * 2);        // 8 MB
  __bf16* wqb   = (__bf16*)alloc((size_t)Dc * Dc * 2);
  __bf16* wkb   = (__bf16*)alloc((size_t)Dc * Dc * 2);
  __bf16* wvb   = (__bf16*)alloc((size_t)Dc * Dc * 2);
  __bf16* wob   = (__bf16*)alloc((size_t)Dc * Dc * 2);
  __bf16* qbuf  = (__bf16*)alloc((size_t)Mr * Dc * 2);        // (b,t,h,dk)
  __bf16* kbuf  = (__bf16*)alloc((size_t)Mr * Dc * 2);
  __bf16* vbuf  = (__bf16*)alloc((size_t)Mr * Dc * 2);
  __bf16* attnb = (__bf16*)alloc((size_t)Mr * Dc * 2);
  __bf16* biasb = (__bf16*)alloc((size_t)BHc * Tc * Tc * 2);  // 128 MB (fits L2)

  // 1) weights -> bf16
  cvt_f32_bf16_kernel<<<1024, 256, 0, stream>>>(Wq, wqb, Dc * Dc);
  cvt_f32_bf16_kernel<<<1024, 256, 0, stream>>>(Wk, wkb, Dc * Dc);
  cvt_f32_bf16_kernel<<<1024, 256, 0, stream>>>(Wv, wvb, Dc * Dc);
  cvt_f32_bf16_kernel<<<1024, 256, 0, stream>>>(Wo, wob, Dc * Dc);

  // 2) LayerNorm -> bf16
  layernorm_kernel<<<Mr, 256, 0, stream>>>(x, ln_g, ln_b, xn);

  // 3) Q/K/V projections  (block tile 128x128)
  dim3 gg(Mr / 128, Dc / 128);
  gemm_bf16_kernel<false><<<gg, 256, 0, stream>>>(xn, wqb, bq, qbuf, Mr, Dc, Dc);
  gemm_bf16_kernel<false><<<gg, 256, 0, stream>>>(xn, wkb, bk, kbuf, Mr, Dc, Dc);
  gemm_bf16_kernel<false><<<gg, 256, 0, stream>>>(xn, wvb, bv, vbuf, Mr, Dc, Dc);

  // 4) relative-position bias: single pass over pos_k (fp32, 256 MB)
  bias_kernel<<<dim3(Tc / 128, Tc), 256, 0, stream>>>(qbuf, pos_k, biasb);

  // 5) flash attention
  attn_kernel<<<dim3(Tc / 128, BHc), 256, 0, stream>>>(qbuf, kbuf, vbuf, biasb, attnb);

  // 6) output projection -> fp32
  gemm_bf16_kernel<true><<<gg, 256, 0, stream>>>(attnb, wob, bo, out, Mr, Dc, Dc);
}
